// Policy_16801912062026
// MI455X (gfx1250) — matrix operand order repaired and checked by reading, not compiled, not run on hardware
//
#include <hip/hip_runtime.h>

typedef __attribute__((ext_vector_type(16))) _Float16 v16h;
typedef __attribute__((ext_vector_type(8)))  _Float16 v8h;
typedef __attribute__((ext_vector_type(8)))  float    v8f;
typedef __attribute__((ext_vector_type(4)))  float    v4f;

#define N_NODES 200000
#define IN_F    500
#define EMB     128
#define HID     64
#define NC      7

// LDS row strides (f16 units), padded so row-to-row advances 4 banks:
//   264*2B = 528B = 132 dw (132%64==4), 136*2B=272B=68 dw (68%64==4),
//   72*2B = 144B = 36 dw (rows n*36 mod 64 all distinct for n<16).
#define SW_EMB_STRIDE 264   // sWemb: [128 n][256 k-half]
#define SW_R1_STRIDE  136   // sWr1 : [64 n][128 k]
#define SW_R2_STRIDE  72    // sWr2 : [16 n][64 k] (n>=7, k>=64 zero)
#define SACT_STRIDE   136   // per-wave activation buffer [16 m][128 k]

__device__ __forceinline__ v16h cat8(v8h lo, v8h hi16) {
  return __builtin_shufflevector(lo, hi16, 0,1,2,3,4,5,6,7,8,9,10,11,12,13,14,15);
}

// B-fragment: element e <-> K = kb*32 + (e<8?e:e+8) + 8*(lane>=16), N = lane&15.
// two 8-f16 runs at +0 and +16 in LDS => two 16B ds_load_b128.
__device__ __forceinline__ v16h load_b_frag(const _Float16* p) {
  v8h b0 = *(const v8h*)(p);
  v8h b1 = *(const v8h*)(p + 16);
  return cat8(b0, b1);
}

// A-fragment from LDS: element e <-> K = kb*32 + (e<8?e:e+8) + 8*(lane>=16), M = lane&15.
__device__ __forceinline__ v16h load_a_lds(const _Float16* base, int stride, int m, int kb, int hi) {
  const _Float16* p = base + m * stride + kb * 32 + hi * 8;
  v8h a0 = *(const v8h*)(p);        // K chunk [+0, +8)
  v8h a1 = *(const v8h*)(p + 16);   // K chunk [+16, +24)
  return cat8(a0, a1);
}

// 8 consecutive K values of one feature row -> elements [e0, e0+8) of A frag.
__device__ __forceinline__ void load_chunk8(v16h& a, int e0, const float* __restrict__ p, int k) {
  if (k + 8 <= IN_F) {
    v4f x0 = *(const v4f*)(p + k);
    v4f x1 = *(const v4f*)(p + k + 4);
    a[e0+0]=(_Float16)x0[0]; a[e0+1]=(_Float16)x0[1]; a[e0+2]=(_Float16)x0[2]; a[e0+3]=(_Float16)x0[3];
    a[e0+4]=(_Float16)x1[0]; a[e0+5]=(_Float16)x1[1]; a[e0+6]=(_Float16)x1[2]; a[e0+7]=(_Float16)x1[3];
  } else {                            // K tail: 500..511 padded with zeros
#pragma unroll
    for (int j = 0; j < 8; ++j)
      a[e0+j] = (k + j < IN_F) ? (_Float16)p[k + j] : (_Float16)0.0f;
  }
}

// Stage W_emb[kbase..kbase+256)[:] transposed to f16 in LDS; zero rows k>=500.
__device__ __forceinline__ void stage_wemb(_Float16* sWemb, const float* __restrict__ Wemb,
                                           int kbase, int tid) {
  const int n4 = (tid & 31) * 4;     // 32 column groups of 4
  const int k0 = tid >> 5;           // 8 k rows per pass
  for (int k = k0; k < 256; k += 8) {
    int kg = kbase + k;
    v4f w = {0.0f, 0.0f, 0.0f, 0.0f};
    if (kg < IN_F) w = *(const v4f*)(Wemb + (long)kg * EMB + n4);
    sWemb[(n4+0)*SW_EMB_STRIDE + k] = (_Float16)w[0];
    sWemb[(n4+1)*SW_EMB_STRIDE + k] = (_Float16)w[1];
    sWemb[(n4+2)*SW_EMB_STRIDE + k] = (_Float16)w[2];
    sWemb[(n4+3)*SW_EMB_STRIDE + k] = (_Float16)w[3];
  }
}

__global__ __launch_bounds__(256, 1)
void mlp3_fused(const float* __restrict__ feat, const float* __restrict__ Wemb,
                const float* __restrict__ bemb, const float* __restrict__ Wr1,
                const float* __restrict__ br1,  const float* __restrict__ Wr2,
                const float* __restrict__ br2,  float* __restrict__ out) {
  __shared__ _Float16 sWemb[EMB * SW_EMB_STRIDE];   // 67,584 B
  __shared__ _Float16 sWr1 [HID * SW_R1_STRIDE];    // 17,408 B
  __shared__ _Float16 sWr2 [16  * SW_R2_STRIDE];    //  2,304 B
  __shared__ _Float16 sAct [8][16 * SACT_STRIDE];   // 34,816 B  (total 122,112 B)

  const int tid  = threadIdx.x;
  const int lane = tid & 31;
  const int wave = tid >> 5;
  const int hi   = lane >> 4;      // lane half selects K sub-chunks
  const int nl   = lane & 15;      // N (B/C) or M (A) within tile

  const long rowbase = (long)blockIdx.x * 128 + (long)wave * 16;
  long arow = rowbase + nl;
  if (arow > (long)(N_NODES - 1)) arow = N_NODES - 1;   // clamp loads; stores guarded
  const float* __restrict__ ap = feat + arow * IN_F;

  // ---- stage W_rt1^T (f16) : sWr1[n][k], n<64, k<128 ----
  {
    const int n4 = (tid & 15) * 4;
    const int k0 = tid >> 4;
    for (int k = k0; k < EMB; k += 16) {
      v4f w = *(const v4f*)(Wr1 + k * HID + n4);
      sWr1[(n4+0)*SW_R1_STRIDE + k] = (_Float16)w[0];
      sWr1[(n4+1)*SW_R1_STRIDE + k] = (_Float16)w[1];
      sWr1[(n4+2)*SW_R1_STRIDE + k] = (_Float16)w[2];
      sWr1[(n4+3)*SW_R1_STRIDE + k] = (_Float16)w[3];
    }
    // ---- stage W_rt2^T zero-padded to N=16 : sWr2[n][k] ----
    for (int i = tid; i < 16 * SW_R2_STRIDE; i += 256) {
      int n = i / SW_R2_STRIDE, k = i % SW_R2_STRIDE;
      float v = (n < NC && k < HID) ? Wr2[k * NC + n] : 0.0f;
      sWr2[i] = (_Float16)v;
    }
  }

  // ================= Layer 1: feat[16x512] @ Wemb[512x128] =================
  stage_wemb(sWemb, Wemb, 0, tid);     // K half [0,256)
  __syncthreads();

  v8f acc[8] = {};
#pragma unroll
  for (int kb = 0; kb < 8; ++kb) {
    v16h a;
    load_chunk8(a, 0, ap, kb * 32 + hi * 8);
    load_chunk8(a, 8, ap, kb * 32 + hi * 8 + 16);
#pragma unroll
    for (int nt = 0; nt < 8; ++nt) {
      v16h b = load_b_frag(&sWemb[(nt * 16 + nl) * SW_EMB_STRIDE + kb * 32 + hi * 8]);
      acc[nt] = __builtin_amdgcn_wmma_f32_16x16x32_f16(false, a, false, b,
                                                       (short)0, acc[nt], false, false);
    }
  }

  __syncthreads();
  stage_wemb(sWemb, Wemb, 256, tid);   // K half [256,512); k>=500 zeroed
  __syncthreads();

#pragma unroll
  for (int kb = 8; kb < 16; ++kb) {
    v16h a;
    load_chunk8(a, 0, ap, kb * 32 + hi * 8);
    load_chunk8(a, 8, ap, kb * 32 + hi * 8 + 16);
    const int kl = (kb - 8) * 32;
#pragma unroll
    for (int nt = 0; nt < 8; ++nt) {
      v16h b = load_b_frag(&sWemb[(nt * 16 + nl) * SW_EMB_STRIDE + kl + hi * 8]);
      acc[nt] = __builtin_amdgcn_wmma_f32_16x16x32_f16(false, a, false, b,
                                                       (short)0, acc[nt], false, false);
    }
  }

  // epilogue 1: bias + ReLU + f16, transpose via per-wave LDS buffer
  _Float16* se = &sAct[wave][0];
#pragma unroll
  for (int nt = 0; nt < 8; ++nt) {
    float bias = bemb[nt * 16 + nl];
#pragma unroll
    for (int r = 0; r < 8; ++r) {
      float v = fmaxf(acc[nt][r] + bias, 0.0f);      // C: M = r + 8*hi, N = nt*16+nl
      se[(r + hi * 8) * SACT_STRIDE + nt * 16 + nl] = (_Float16)v;
    }
  }

  // ================= Layer 2: emb[16x128] @ Wr1[128x64] =================
  v8f hacc[4] = {};
#pragma unroll
  for (int kb = 0; kb < 4; ++kb) {
    v16h a = load_a_lds(se, SACT_STRIDE, nl, kb, hi);
#pragma unroll
    for (int nt = 0; nt < 4; ++nt) {
      v16h b = load_b_frag(&sWr1[(nt * 16 + nl) * SW_R1_STRIDE + kb * 32 + hi * 8]);
      hacc[nt] = __builtin_amdgcn_wmma_f32_16x16x32_f16(false, a, false, b,
                                                        (short)0, hacc[nt], false, false);
    }
  }

  // epilogue 2: bias + ReLU + f16 into same per-wave buffer (reads done)
#pragma unroll
  for (int nt = 0; nt < 4; ++nt) {
    float bias = br1[nt * 16 + nl];
#pragma unroll
    for (int r = 0; r < 8; ++r) {
      float v = fmaxf(hacc[nt][r] + bias, 0.0f);
      se[(r + hi * 8) * SW_R2_STRIDE + nt * 16 + nl] = (_Float16)v;
    }
  }

  // ================= Layer 3: h[16x64] @ Wr2[64x16(pad)] =================
  v8f oacc = {};
#pragma unroll
  for (int kb = 0; kb < 2; ++kb) {
    v16h a = load_a_lds(se, SW_R2_STRIDE, nl, kb, hi);
    v16h b = load_b_frag(&sWr2[nl * SW_R2_STRIDE + kb * 32 + hi * 8]);
    oacc = __builtin_amdgcn_wmma_f32_16x16x32_f16(false, a, false, b,
                                                  (short)0, oacc, false, false);
  }

  if (nl < NC) {
    float bias = br2[nl];
#pragma unroll
    for (int r = 0; r < 8; ++r) {
      long row = rowbase + r + hi * 8;
      if (row < (long)N_NODES) out[row * NC + nl] = oacc[r] + bias;
    }
  }
}

extern "C" void kernel_launch(void* const* d_in, const int* in_sizes, int n_in,
                              void* d_out, int out_size, void* d_ws, size_t ws_size,
                              hipStream_t stream) {
  (void)in_sizes; (void)n_in; (void)out_size; (void)d_ws; (void)ws_size;
  const float* feat = (const float*)d_in[1];   // d_in[0] = adj (unused)
  const float* Wemb = (const float*)d_in[2];
  const float* bemb = (const float*)d_in[3];
  const float* Wr1  = (const float*)d_in[4];
  const float* br1  = (const float*)d_in[5];
  const float* Wr2  = (const float*)d_in[6];
  const float* br2  = (const float*)d_in[7];
  float* out = (float*)d_out;

  const int nblocks = (N_NODES + 127) / 128;   // 1563
  hipLaunchKernelGGL(mlp3_fused, dim3(nblocks), dim3(256), 0, stream,
                     feat, Wemb, bemb, Wr1, br1, Wr2, br2, out);
}
